// PoseFusion_7310034338273
// MI455X (gfx1250) — matrix operand order repaired and checked by reading, not compiled, not run on hardware
//
#include <hip/hip_runtime.h>
#include <hip/hip_bf16.h>

// ---------------------------------------------------------------------------
// PoseFusion for MI455X (gfx1250, wave32).
//  * All GEMM FLOPs on v_wmma_f32_16x16x32_f16 (f32 accumulate).
//  * A/B operands pre-swizzled into WMMA lane-fragment order so each lane
//    loads its 16-half fragment as one contiguous 32B (b128-pair) load.
//  * Edge segment-softmax / layernorms / top-k in f32 VALU + float atomics.
// ---------------------------------------------------------------------------

#define B_    16
#define N_    1024
#define CIN_  64
#define H_    8
#define F_    512
#define DEG_  8
#define NB_   (B_ * N_)      // 16384
#define E_    (NB_ * DEG_)   // 131072
#define NCLS_ 100
#define NCLSP_ 112           // cls cols padded to 7 tiles of 16
#define KPOOL_ 256
#define NOUT1_ 2048          // Q | K | V | skip fused output width
#define QKVW_  1536          // stored width of Q|K|V buffer
#define NT1_   128           // NOUT1_/16 tiles
#define MT2_   256           // (B_*KPOOL_)/16
#define KT2_   16            // F_/32
#define LNEPS 1e-5f

typedef __attribute__((ext_vector_type(16))) _Float16 v16h;
typedef __attribute__((ext_vector_type(8)))  float    v8f;

// ---------------------------- small helpers --------------------------------

__device__ __forceinline__ void atomicMaxF(float* addr, float val) {
    int* ia = (int*)addr;
    float old = __int_as_float(__hip_atomic_load(ia, __ATOMIC_RELAXED, __HIP_MEMORY_SCOPE_AGENT));
    while (val > old) {
        int assumed = __float_as_int(old);
        int prev = atomicCAS(ia, assumed, __float_as_int(val));
        if (prev == assumed) break;
        old = __int_as_float(prev);
    }
}

__device__ __forceinline__ float block_reduce_sum_256(float v, float* red) {
    int t = threadIdx.x;
    red[t] = v; __syncthreads();
    for (int s = 128; s > 0; s >>= 1) {
        if (t < s) red[t] += red[t + s];
        __syncthreads();
    }
    float r = red[0]; __syncthreads();
    return r;
}

// fragment position -> channel offset within a 32-wide K slab (16-bit A layout)
__device__ __forceinline__ int a_frag_ch(int hi, int pos) {
    // pos 0..7 -> K = hi*8 + pos ; pos 8..15 -> K = 16 + hi*8 + (pos-8)
    return (pos < 8) ? (hi * 8 + pos) : (16 + hi * 8 + (pos - 8));
}

// ------------------------ swizzling conversions ----------------------------

// X (16384x64 f32) -> A-fragment order: Xs[((mt*32+lane)*32) + p]
// p in [0,16): a0 fragment element p (K slab 0); p in [16,32): a1 (K slab 32)
__global__ void k_cvt_x_swz(const float* __restrict__ x, _Float16* __restrict__ Xs) {
    int i = blockIdx.x * blockDim.x + threadIdx.x;      // NB_*CIN_ = 1048576
    if (i >= NB_ * CIN_) return;
    int p    = i & 31;
    int lane = (i >> 5) & 31;
    int mt   = i >> 10;
    int hi   = lane >> 4;
    int frag = p >> 4;
    int pos  = p & 15;
    int ch   = frag * 32 + a_frag_ch(hi, pos);
    int row  = mt * 16 + (lane & 15);
    Xs[i] = (_Float16)x[(long)row * CIN_ + ch];
}

// [Wq|Wk|Wv|Wskip] -> B-fragment order: W1s[(((kt*NT1_)+nt)*32+lane)*16 + j]
// fragment element j of lane: W[kt*32 + a_frag_ch(lane>>4, j)][nt*16 + (lane&15)]
__global__ void k_build_w1_swz(const float* __restrict__ Wq, const float* __restrict__ Wk,
                               const float* __restrict__ Wv, const float* __restrict__ Wsk,
                               _Float16* __restrict__ W1s) {
    int i = blockIdx.x * blockDim.x + threadIdx.x;      // 2*NT1_*32*16 = 131072
    if (i >= CIN_ * NOUT1_) return;
    int j    = i & 15;
    int lane = (i >> 4) & 31;
    int nt   = (i >> 9) & (NT1_ - 1);
    int kt   = i >> 16;
    int kg   = kt * 32 + a_frag_ch(lane >> 4, j);
    int col  = nt * 16 + (lane & 15);
    int sel  = col >> 9, c = col & (F_ - 1);
    const float* W = (sel == 0) ? Wq : (sel == 1) ? Wk : (sel == 2) ? Wv : Wsk;
    W1s[i] = (_Float16)W[kg * F_ + c];
}

__global__ void k_build_b1(const float* __restrict__ bq, const float* __restrict__ bk,
                           const float* __restrict__ bv, const float* __restrict__ bs,
                           float* __restrict__ b1) {
    int j = blockIdx.x * blockDim.x + threadIdx.x;
    if (j >= NOUT1_) return;
    int sel = j / F_, c = j % F_;
    b1[j] = (sel == 0) ? bq[c] : (sel == 1) ? bk[c] : (sel == 2) ? bv[c] : bs[c];
}

// cls_w (512x100, pad to 112) then mlp_w (512x64), each in B-fragment order.
__global__ void k_build_w2_swz(const float* __restrict__ clsw, const float* __restrict__ mlpw,
                               _Float16* __restrict__ W2s) {
    int i = blockIdx.x * blockDim.x + threadIdx.x;
    const int n_cls = KT2_ * 7 * 32 * 16;               // 57344
    const int n_tot = n_cls + KT2_ * 4 * 32 * 16;       // + 32768
    if (i >= n_tot) return;
    if (i < n_cls) {
        int j = i & 15, lane = (i >> 4) & 31;
        int nt = (i / 512) % 7, kt = i / 3584;
        int kg  = kt * 32 + a_frag_ch(lane >> 4, j);
        int col = nt * 16 + (lane & 15);
        W2s[i] = (col < NCLS_) ? (_Float16)clsw[kg * NCLS_ + col] : (_Float16)0.0f;
    } else {
        int i2 = i - n_cls;
        int j = i2 & 15, lane = (i2 >> 4) & 31;
        int nt = (i2 / 512) % 4, kt = i2 / 2048;
        int kg  = kt * 32 + a_frag_ch(lane >> 4, j);
        int col = nt * 16 + (lane & 15);
        W2s[i] = (_Float16)mlpw[kg * CIN_ + col];
    }
}

__global__ void k_inv_norm(const float* __restrict__ pw, float* __restrict__ invn) {
    __shared__ float red[256];
    int t = threadIdx.x;
    float v = pw[t] * pw[t] + pw[t + 256] * pw[t + 256];
    float s = block_reduce_sum_256(v, red);
    if (t == 0) invn[0] = rsqrtf(s);
}

__global__ void k_init_softmax(float* __restrict__ amax, float* __restrict__ den) {
    int i = blockIdx.x * blockDim.x + threadIdx.x;
    if (i < NB_ * H_) { amax[i] = -3.4e38f; den[i] = 0.0f; }
}

// ------------------------------ GEMM 1 -------------------------------------
// Xs @ W1s + b1 -> Q|K|V into qkv (f32, width 1536); skip cols seed outacc.
// One wave: 16 rows x 64 cols (A reused over 4 N-tiles, 2 WMMAs each, K=64).
__global__ void k_gemm1(const _Float16* __restrict__ Xs, const _Float16* __restrict__ W1s,
                        const float* __restrict__ b1,
                        float* __restrict__ qkv, float* __restrict__ outacc) {
    int wave = threadIdx.x >> 5;
    int lane = threadIdx.x & 31;
    int task = blockIdx.x * 8 + wave;          // 1024 mtiles * 32 nquads
    int mtile = task >> 5;
    int nquad = task & 31;
    int hi = lane >> 4;
    int lrow = lane & 15;

    const v16h* ap = (const v16h*)(Xs + ((long)mtile * 32 + lane) * 32);
    v16h a0 = ap[0];
    v16h a1 = ap[1];
    const v16h* bp = (const v16h*)W1s;

#pragma unroll
    for (int t = 0; t < 4; ++t) {
        int ntile = nquad * 4 + t;
        int colbase = ntile * 16;              // tile-uniform
        int col = colbase + lrow;
        v16h bf0 = bp[((long)ntile) * 32 + lane];
        v16h bf1 = bp[((long)NT1_ + ntile) * 32 + lane];
        float bias = b1[col];
        v8f c;
#pragma unroll
        for (int j = 0; j < 8; ++j) c[j] = bias;
        c = __builtin_amdgcn_wmma_f32_16x16x32_f16(false, a0, false, bf0, (short)0, c, false, false);
        c = __builtin_amdgcn_wmma_f32_16x16x32_f16(false, a1, false, bf1, (short)0, c, false, false);
        if (colbase < QKVW_) {                 // uniform branch: Q|K|V region
#pragma unroll
            for (int j = 0; j < 8; ++j)
                qkv[(long)(mtile * 16 + j + hi * 8) * QKVW_ + col] = c[j];
        } else {                               // uniform branch: skip region
#pragma unroll
            for (int j = 0; j < 8; ++j)
                outacc[(long)(mtile * 16 + j + hi * 8) * F_ + (col - QKVW_)] = c[j];
        }
    }
}

// --------------------------- edge attention --------------------------------

__global__ void k_edge_alpha(const float* __restrict__ qkv, const int* __restrict__ ei,
                             float* __restrict__ alpha, float* __restrict__ amax) {
    int tid = blockIdx.x * blockDim.x + threadIdx.x;    // E_*H_
    if (tid >= E_ * H_) return;
    int e = tid >> 3, h = tid & 7;
    int s = ei[e], d = ei[E_ + e];
    const float* qr = qkv + (long)d * QKVW_ + h * CIN_;          // Q cols 0..511
    const float* kr = qkv + (long)s * QKVW_ + F_ + h * CIN_;     // K cols 512..1023
    float acc = 0.0f;
#pragma unroll 8
    for (int c = 0; c < CIN_; ++c) acc += qr[c] * kr[c];
    acc *= 0.125f;                                               // 1/sqrt(64)
    alpha[tid] = acc;
    atomicMaxF(&amax[d * H_ + h], acc);
}

__global__ void k_edge_exp(const int* __restrict__ ei, float* __restrict__ alpha,
                           const float* __restrict__ amax, float* __restrict__ den) {
    int tid = blockIdx.x * blockDim.x + threadIdx.x;
    if (tid >= E_ * H_) return;
    int e = tid >> 3, h = tid & 7;
    int d = ei[E_ + e];
    float ex = __expf(alpha[tid] - amax[d * H_ + h]);
    alpha[tid] = ex;
    atomicAdd(&den[d * H_ + h], ex);
}

__global__ void k_edge_accum(const float* __restrict__ qkv, const int* __restrict__ ei,
                             const float* __restrict__ alpha, const float* __restrict__ den,
                             float* __restrict__ outacc) {
    int e = blockIdx.x;
    int s = ei[e], d = ei[E_ + e];
    const float* vr = qkv + (long)s * QKVW_ + 2 * F_;            // V cols 1024..1535
#pragma unroll
    for (int c = threadIdx.x; c < F_; c += 256) {
        int h = c >> 6;
        float w = alpha[e * H_ + h] / den[d * H_ + h];
        atomicAdd(&outacc[(long)d * F_ + c], w * vr[c]);
    }
}

// ----------------- LN1 + ReLU + pooling score (fused) ----------------------

__global__ void k_ln1_score(float* __restrict__ x /*in/out 16384x512*/,
                            const float* __restrict__ nw, const float* __restrict__ nb,
                            const float* __restrict__ pw, const float* __restrict__ invn,
                            float* __restrict__ score) {
    __shared__ float red[256];
    long row = blockIdx.x;
    int t = threadIdx.x;
    float x0 = x[row * F_ + t], x1 = x[row * F_ + 256 + t];
    float mean = block_reduce_sum_256(x0 + x1, red) * (1.0f / F_);
    float d0 = x0 - mean, d1 = x1 - mean;
    float var = block_reduce_sum_256(d0 * d0 + d1 * d1, red) * (1.0f / F_);
    float rs = rsqrtf(var + LNEPS);
    float y0 = fmaxf(d0 * rs * nw[t] + nb[t], 0.0f);
    float y1 = fmaxf(d1 * rs * nw[t + 256] + nb[t + 256], 0.0f);
    x[row * F_ + t] = y0;
    x[row * F_ + 256 + t] = y1;
    float sdot = block_reduce_sum_256(y0 * pw[t] + y1 * pw[t + 256], red);
    if (t == 0) score[row] = sdot * invn[0];
}

// ------------------------------- top-k -------------------------------------
// rank-based selection, matches jax.lax.top_k (desc, stable low-index ties)
__global__ void k_topk(const float* __restrict__ score, int* __restrict__ selidx,
                       float* __restrict__ selscore) {
    __shared__ float s[N_];
    int b = blockIdx.x;
    int i = threadIdx.x;
    float v = score[b * N_ + i];
    s[i] = v;
    __syncthreads();
    int rank = 0;
    for (int j = 0; j < N_; ++j) {
        float u = s[j];
        rank += (u > v) || (u == v && j < i);
    }
    if (rank < KPOOL_) {
        selidx[b * KPOOL_ + rank] = i;
        selscore[b * KPOOL_ + rank] = v;
    }
}

// ---- gather + tanh-scale + LN2 + ReLU -> pooled in A-fragment f16 order ----

__device__ __forceinline__ long pooled_swz_idx(int mt, int lrow, int c) {
    int kt = c >> 5;
    int rem = c & 31;
    int hi, pos;
    if (rem < 16) { hi = rem >> 3; pos = rem & 7; }
    else          { int r2 = rem - 16; hi = r2 >> 3; pos = (r2 & 7) + 8; }
    int lane = lrow + hi * 16;
    return (((long)mt * KT2_ + kt) * 32 + lane) * 16 + pos;
}

__global__ void k_gather_ln2(const float* __restrict__ pf, const int* __restrict__ selidx,
                             const float* __restrict__ selscore,
                             const float* __restrict__ nw, const float* __restrict__ nb,
                             _Float16* __restrict__ pooleds) {
    __shared__ float red[256];
    int r = blockIdx.x;                       // pooled row, 0..4095
    int b = r >> 8;
    int t = threadIdx.x;
    long srow = (long)b * N_ + selidx[r];
    float sc = tanhf(selscore[r]);
    float x0 = pf[srow * F_ + t] * sc, x1 = pf[srow * F_ + 256 + t] * sc;
    float mean = block_reduce_sum_256(x0 + x1, red) * (1.0f / F_);
    float d0 = x0 - mean, d1 = x1 - mean;
    float var = block_reduce_sum_256(d0 * d0 + d1 * d1, red) * (1.0f / F_);
    float rs = rsqrtf(var + LNEPS);
    float y0 = fmaxf(d0 * rs * nw[t] + nb[t], 0.0f);
    float y1 = fmaxf(d1 * rs * nw[t + 256] + nb[t + 256], 0.0f);
    int mt = r >> 4, lrow = r & 15;
    pooleds[pooled_swz_idx(mt, lrow, t)]       = (_Float16)y0;
    pooleds[pooled_swz_idx(mt, lrow, t + 256)] = (_Float16)y1;
}

// ------------------------------ GEMM 2 -------------------------------------
// pooled(4096x512,f16 swizzled) @ W2(512xN,f16 swizzled) + bias -> d_out.
// K = 512 -> 16 chained v_wmma_f32_16x16x32_f16 per 16x16 tile.
__global__ void k_gemm2(const _Float16* __restrict__ As, const _Float16* __restrict__ Wsw,
                        const float* __restrict__ bias, float* __restrict__ out,
                        int ncols, int ntiles) {
    int wave = threadIdx.x >> 5;
    int lane = threadIdx.x & 31;
    int task = blockIdx.x * 8 + wave;          // MT2_ * ntiles
    int mtile = task / ntiles;
    int ntile = task % ntiles;
    int hi = lane >> 4;
    int lrow = lane & 15;
    int col = ntile * 16 + lrow;

    const v16h* ap = (const v16h*)As;
    const v16h* bp = (const v16h*)Wsw;

    float bias_c = (col < ncols) ? bias[col] : 0.0f;
    v8f c;
#pragma unroll
    for (int j = 0; j < 8; ++j) c[j] = bias_c;

#pragma unroll
    for (int kt = 0; kt < KT2_; ++kt) {
        v16h a  = ap[((long)mtile * KT2_ + kt) * 32 + lane];
        v16h bf = bp[((long)kt * ntiles + ntile) * 32 + lane];
        c = __builtin_amdgcn_wmma_f32_16x16x32_f16(false, a, false, bf, (short)0, c, false, false);
    }
    if (col < ncols) {
#pragma unroll
        for (int j = 0; j < 8; ++j)
            out[(long)(mtile * 16 + j + hi * 8) * ncols + col] = c[j];
    }
}

// ------------------------------ launcher -----------------------------------

extern "C" void kernel_launch(void* const* d_in, const int* in_sizes, int n_in,
                              void* d_out, int out_size, void* d_ws, size_t ws_size,
                              hipStream_t stream) {
    const float* pose  = (const float*)d_in[0];
    const float* Wq    = (const float*)d_in[1];  const float* bq = (const float*)d_in[2];
    const float* Wk    = (const float*)d_in[3];  const float* bk = (const float*)d_in[4];
    const float* Wv    = (const float*)d_in[5];  const float* bv = (const float*)d_in[6];
    const float* Wsk   = (const float*)d_in[7];  const float* bs = (const float*)d_in[8];
    const float* n1w   = (const float*)d_in[9];  const float* n1b = (const float*)d_in[10];
    const float* poolw = (const float*)d_in[11];
    const float* n2w   = (const float*)d_in[12]; const float* n2b = (const float*)d_in[13];
    const float* mlpw  = (const float*)d_in[14]; const float* mlpb = (const float*)d_in[15];
    const float* clsw  = (const float*)d_in[16]; const float* clsb = (const float*)d_in[17];
    const int*   eidx  = (const int*)d_in[18];
    (void)in_sizes; (void)n_in; (void)out_size; (void)ws_size;

    // ---- carve scratch (256B aligned) ----
    char* base = (char*)d_ws;
    size_t off = 0;
    auto carve = [&](size_t bytes) -> void* {
        void* p = base + off;
        off = (off + bytes + 255) & ~(size_t)255;
        return p;
    };
    _Float16* Xs      = (_Float16*)carve((size_t)NB_ * CIN_ * 2);
    _Float16* W1s     = (_Float16*)carve((size_t)CIN_ * NOUT1_ * 2);
    float*    b1      = (float*)   carve((size_t)NOUT1_ * 4);
    float*    qkv     = (float*)   carve((size_t)NB_ * QKVW_ * 4);
    float*    outacc  = (float*)   carve((size_t)NB_ * F_ * 4);     // skip+attn, then pf
    float*    alpha   = (float*)   carve((size_t)E_ * H_ * 4);
    float*    amax    = (float*)   carve((size_t)NB_ * H_ * 4);
    float*    den     = (float*)   carve((size_t)NB_ * H_ * 4);
    float*    score   = (float*)   carve((size_t)NB_ * 4);
    int*      selidx  = (int*)     carve((size_t)B_ * KPOOL_ * 4);
    float*    selscore= (float*)   carve((size_t)B_ * KPOOL_ * 4);
    _Float16* pooleds = (_Float16*)carve((size_t)B_ * KPOOL_ * F_ * 2);
    _Float16* W2s     = (_Float16*)carve((size_t)F_ * (NCLSP_ + CIN_) * 2);
    float*    invn    = (float*)   carve(256);

    // ---- prep / swizzled conversions ----
    k_cvt_x_swz  <<<(NB_ * CIN_ + 255) / 256, 256, 0, stream>>>(pose, Xs);
    k_build_w1_swz<<<(CIN_ * NOUT1_ + 255) / 256, 256, 0, stream>>>(Wq, Wk, Wv, Wsk, W1s);
    k_build_b1   <<<(NOUT1_ + 255) / 256, 256, 0, stream>>>(bq, bk, bv, bs, b1);
    k_build_w2_swz<<<(F_ * (NCLSP_ + CIN_) + 255) / 256, 256, 0, stream>>>(clsw, mlpw, W2s);
    k_inv_norm   <<<1, 256, 0, stream>>>(poolw, invn);
    k_init_softmax<<<(NB_ * H_ + 255) / 256, 256, 0, stream>>>(amax, den);

    // ---- fused QKV+skip GEMM (WMMA) ----
    k_gemm1<<<(1024 * 32) / 8, 256, 0, stream>>>(Xs, W1s, b1, qkv, outacc);

    // ---- edge attention (segment softmax + scatter) ----
    k_edge_alpha<<<(E_ * H_ + 255) / 256, 256, 0, stream>>>(qkv, eidx, alpha, amax);
    k_edge_exp  <<<(E_ * H_ + 255) / 256, 256, 0, stream>>>(eidx, alpha, amax, den);
    k_edge_accum<<<E_, 256, 0, stream>>>(qkv, eidx, alpha, den, outacc);

    // ---- LN1 + ReLU + pool score ----
    k_ln1_score<<<NB_, 256, 0, stream>>>(outacc, n1w, n1b, poolw, invn, score);

    // ---- top-k per graph ----
    k_topk<<<B_, N_, 0, stream>>>(score, selidx, selscore);

    // ---- gather + tanh scale + LN2 + ReLU -> swizzled f16 ----
    k_gather_ln2<<<B_ * KPOOL_, 256, 0, stream>>>(outacc, selidx, selscore, n2w, n2b, pooleds);

    // ---- heads (WMMA): cls (4096x100) then op (4096x64), concat in d_out ----
    float* out_cls = (float*)d_out;
    float* out_op  = out_cls + (size_t)(B_ * KPOOL_) * NCLS_;
    k_gemm2<<<(MT2_ * 7) / 8, 256, 0, stream>>>(pooleds, W2s, clsb, out_cls, NCLS_, 7);
    k_gemm2<<<(MT2_ * 4) / 8, 256, 0, stream>>>(pooleds, W2s + (size_t)F_ * NCLSP_, mlpb,
                                                out_op, CIN_, 4);
}
